// LocalConvPhasorBlock_45337674776747
// MI455X (gfx1250) — compile-verified, matrix operand order checked by reading, not run on hardware
//
#include <hip/hip_runtime.h>

// ---------------------------------------------------------------------------
// LocalConvPhasorBlock for MI455X (gfx1250, wave32, WMMA).
// All dense projections run through a bf16 v_wmma_f32_16x16x32_bf16 GEMM with
// fused bias/addend/activation. LDS tiles are stored pre-swizzled in the
// WMMA fragment layout so fragment reads are pure ds_load_b128 pairs.
// Sequential cumsums are parallelized as chunked scans.
// Workspace requirement: ~790 MB (see offset map below).
// ---------------------------------------------------------------------------

typedef __bf16           v16bf __attribute__((ext_vector_type(16)));
typedef __bf16           v2bf  __attribute__((ext_vector_type(2)));
typedef unsigned int     v8ui  __attribute__((ext_vector_type(8)));
typedef float            v8f   __attribute__((ext_vector_type(8)));

#define PI_F 3.14159265358979323846f

// ---------------- problem constants ----------------
constexpr int    Bx = 8, Lx = 4096, Dx = 512, Px = 128, Vx = 8;
constexpr int    NC = 32, CL = 128;                 // KV scan: 32 chunks of 128
constexpr size_t Tt = (size_t)Bx * Lx;              // 32768 tokens
constexpr size_t TD = Tt * Dx;                      // 16,777,216 floats

// ---------------- workspace offset map (floats) ----------------
constexpr size_t O_CG    = 0;                       // conv*gate      (T,D)
constexpr size_t O_CTX   = TD;                      // context_avg    (T,D)
constexpr size_t O_V1    = 2 * TD;                  // v1             (T,D)
constexpr size_t O_MAG   = 3 * TD;                  // magnitude      (T,D)
constexpr size_t O_OFF   = 4 * TD;                  // offset (phase) (T,D)
constexpr size_t O_POSO  = 5 * TD;                  // posoff         (L,D)
constexpr size_t O_QP    = O_POSO + (size_t)Lx * Dx;// query_phase    (T,P)
constexpr size_t O_SP    = O_QP + Tt * Px;          // storage_phase  (T,P)
constexpr size_t O_SKX   = O_SP + Tt * Px;          // x@W_sk1_top    (T,D)
constexpr size_t O_SKH   = O_SKX + TD;              // sk_h           (T,D)
constexpr size_t O_PRET  = O_SKH + TD;              // pos_ret        (T,D)
constexpr size_t O_POUT  = O_PRET + TD;             // pos_out        (T,D)
constexpr size_t O_CONV  = O_POUT + TD;             // conv_out       (T,D)
constexpr size_t O_KVO   = O_CONV + TD;             // kv_out         (T,D)
constexpr size_t O_VALS  = O_KVO + TD;              // values         (T,V)
constexpr size_t O_SG    = O_VALS + Tt * Vx;        // store_gate     (T)
constexpr size_t O_CCC   = O_SG + Tt;               // chunk cos sums (B,NC,P,V)
constexpr size_t O_CCS   = O_CCC + (size_t)Bx * NC * Px * Vx;
constexpr size_t O_CGT   = O_CCS + (size_t)Bx * NC * Px * Vx; // chunk gates
constexpr size_t O_KVR   = O_CGT + (size_t)Bx * NC; // kv_retrieved   (T,V)
constexpr size_t O_COMB  = 2 * TD;                  // LN out (T,3D) aliases v1/mag/off (dead)
constexpr size_t O_H1    = 0;                       // gelu hidden (T,2D) aliases cg/ctx (dead)

// ---------------- helpers ----------------
__device__ __forceinline__ unsigned int f2bf(float f) {     // RNE fp32->bf16 bits
  unsigned int u = __float_as_uint(f);
  u += 0x7FFFu + ((u >> 16) & 1u);
  return u >> 16;
}
__device__ __forceinline__ unsigned int pack_bf(float lo, float hi) {
#if __has_builtin(__builtin_amdgcn_cvt_pk_bf16_f32)
  // single v_cvt_pk_bf16_f32 when the toolchain exposes it
  v2bf p = __builtin_amdgcn_cvt_pk_bf16_f32(lo, hi);
  return __builtin_bit_cast(unsigned int, p);
#else
  return f2bf(lo) | (f2bf(hi) << 16);
#endif
}
__device__ __forceinline__ float act_apply(float v, int act) {
  switch (act) {
    case 1:  return 1.0f / (1.0f + __expf(-v));            // sigmoid
    case 2:  return tanhf(v) * PI_F;                       // tanh * pi
    case 3:  return 0.5f * v * (1.0f + erff(v * 0.7071067811865476f)); // exact gelu
    default: return v;
  }
}

// ---------------------------------------------------------------------------
// Generic GEMM: C[M,N] = act(A[M,K] @ W[K,N] + bias + addin).  M%64==0,
// N%64==0, K%32==0 (true for every call site).  64x64x32 tiles, 4 waves,
// each wave computes a 32x32 sub-tile via 4 bf16 WMMAs per K-step.
//
// LDS holds tiles PRE-SWIZZLED into the CDNA5 16-bit fragment layout:
//   slot [frag][lane][j] (8 dwords = 16 bf16 per (frag,lane))
//   A: frag f covers rows f*16..f*16+15 ; lane half owns K {kb..kb+7, kb+16..kb+23}
//   B: frag f covers cols f*16..f*16+15 ; same K split.
// A fragment read is then one 32B vector load -> two ds_load_b128.
// addin row = gr & addin_mask (mask = L-1 for per-position broadcast,
// 0x7FFFFFFF for identity).
// ---------------------------------------------------------------------------
__global__ __launch_bounds__(128)
void k_gemm_wmma(const float* __restrict__ A, const float* __restrict__ W,
                 const float* __restrict__ bias, const float* __restrict__ addin,
                 int addin_mask, float* __restrict__ C,
                 int M, int Kd, int N, int act)
{
  __shared__ unsigned int AsX[4][32][8];
  __shared__ unsigned int BsX[4][32][8];

  const int tid  = threadIdx.x;
  const int wave = tid >> 5;
  const int lane = tid & 31;
  const int wm   = wave >> 1, wn = wave & 1;
  const int m0   = blockIdx.y * 64, n0 = blockIdx.x * 64;
  const int half = lane >> 4;
  const int l15  = lane & 15;
  const int kb   = half * 8;
  const int f    = wave;                       // staging slot owned by this wave

  v8f acc[2][2];
  #pragma unroll
  for (int i = 0; i < 2; i++)
    #pragma unroll
    for (int j = 0; j < 2; j++) {
      v8f z = {0.f, 0.f, 0.f, 0.f, 0.f, 0.f, 0.f, 0.f};
      acc[i][j] = z;
    }

  // this thread's staging sources
  const float* Arow = A + (size_t)(m0 + f * 16 + l15) * Kd + kb; // + k0 in loop
  const int    ncol = n0 + f * 16 + l15;

  for (int k0 = 0; k0 < Kd; k0 += 32) {
    // ---- stage A: two runs of 8 consecutive floats (32B aligned) ----
    v8ui pa;
    #pragma unroll
    for (int j = 0; j < 4; j++) {
      pa[j]     = pack_bf(Arow[k0 + 2 * j],      Arow[k0 + 2 * j + 1]);
      pa[4 + j] = pack_bf(Arow[k0 + 16 + 2 * j], Arow[k0 + 16 + 2 * j + 1]);
    }
    *(v8ui*)&AsX[f][lane][0] = pa;

    // ---- stage B: strided-by-N loads, coalesced across lanes ----
    v8ui pb;
    #pragma unroll
    for (int j = 0; j < 4; j++) {
      pb[j]     = pack_bf(W[(size_t)(k0 + kb + 2 * j) * N + ncol],
                          W[(size_t)(k0 + kb + 2 * j + 1) * N + ncol]);
      pb[4 + j] = pack_bf(W[(size_t)(k0 + kb + 16 + 2 * j) * N + ncol],
                          W[(size_t)(k0 + kb + 16 + 2 * j + 1) * N + ncol]);
    }
    *(v8ui*)&BsX[f][lane][0] = pb;

    if (k0 + 32 < Kd) {   // pull next tiles toward cache (global_prefetch_b8)
      __builtin_prefetch(&Arow[k0 + 32], 0, 1);
      __builtin_prefetch(&W[(size_t)(k0 + 32 + kb) * N + ncol], 0, 1);
    }
    __syncthreads();

    // ---- fragment reads: pure 32B vector loads (2x ds_load_b128 each) ----
    v16bf a0 = __builtin_bit_cast(v16bf, *(const v8ui*)&AsX[wm * 2 + 0][lane][0]);
    v16bf a1 = __builtin_bit_cast(v16bf, *(const v8ui*)&AsX[wm * 2 + 1][lane][0]);
    v16bf b0 = __builtin_bit_cast(v16bf, *(const v8ui*)&BsX[wn * 2 + 0][lane][0]);
    v16bf b1 = __builtin_bit_cast(v16bf, *(const v8ui*)&BsX[wn * 2 + 1][lane][0]);

    acc[0][0] = __builtin_amdgcn_wmma_f32_16x16x32_bf16(false, a0, false, b0, (short)0, acc[0][0], false, false);
    acc[0][1] = __builtin_amdgcn_wmma_f32_16x16x32_bf16(false, a0, false, b1, (short)0, acc[0][1], false, false);
    acc[1][0] = __builtin_amdgcn_wmma_f32_16x16x32_bf16(false, a1, false, b0, (short)0, acc[1][0], false, false);
    acc[1][1] = __builtin_amdgcn_wmma_f32_16x16x32_bf16(false, a1, false, b1, (short)0, acc[1][1], false, false);
    __syncthreads();
  }

  // epilogue: C layout = lanes 0-15 rows 0..7, lanes 16-31 rows 8..15
  #pragma unroll
  for (int am = 0; am < 2; am++) {
    #pragma unroll
    for (int bn = 0; bn < 2; bn++) {
      int gc = n0 + wn * 32 + bn * 16 + l15;
      float bv = bias ? bias[gc] : 0.0f;
      #pragma unroll
      for (int i = 0; i < 8; i++) {
        int gr = m0 + wm * 32 + am * 16 + half * 8 + i;
        float v = acc[am][bn][i] + bv;
        if (addin) v += addin[(size_t)(gr & addin_mask) * N + gc];
        C[(size_t)gr * N + gc] = act_apply(v, act);
      }
    }
  }
}

// ---------------------------------------------------------------------------
// Causal depthwise conv (K=4) + sigmoid gate, fused: cg = (conv+b)*sig(convg+b)
// ---------------------------------------------------------------------------
__global__ __launch_bounds__(256)
void k_dwconv(const float* __restrict__ x, const float* __restrict__ wl,
              const float* __restrict__ bl, const float* __restrict__ wg,
              const float* __restrict__ bg, float* __restrict__ cg)
{
  size_t gid = (size_t)blockIdx.x * 256 + threadIdx.x;   // t*512 + d
  size_t t = gid >> 9;
  int d = (int)(gid & 511);
  int l = (int)(t & (size_t)(Lx - 1));
  size_t brow = t - l;                                    // b*L
  float a1 = bl[d], a2 = bg[d];
  #pragma unroll
  for (int j = 0; j < 4; j++) {
    int ls = l - 3 + j;
    if (ls >= 0) {
      float xv = x[(brow + ls) * 512 + d];
      a1 += xv * wl[d * 4 + j];
      a2 += xv * wg[d * 4 + j];
    }
  }
  cg[gid] = a1 * (1.0f / (1.0f + __expf(-a2)));
}

// context_avg[b,l,d] = cumsum_l(x)/(l+1) ; one thread per (b,d)
__global__ __launch_bounds__(256)
void k_ctx_scan(const float* __restrict__ x, float* __restrict__ ctx)
{
  int gid = blockIdx.x * 256 + threadIdx.x;   // 0..B*D-1
  int b = gid >> 9, d = gid & 511;
  float s = 0.0f;
  for (int l = 0; l < Lx; l++) {
    size_t t = (size_t)b * Lx + l;
    s += x[t * 512 + d];
    ctx[t * 512 + d] = s / (float)(l + 1);
  }
}

// values (V=8) and store_gate: too narrow for WMMA -> VALU dot products
__global__ __launch_bounds__(256)
void k_valsg(const float* __restrict__ x, const float* __restrict__ Wv,
             const float* __restrict__ bv, const float* __restrict__ Wsg,
             const float* __restrict__ bsg, float* __restrict__ vals,
             float* __restrict__ sg)
{
  size_t gid = (size_t)blockIdx.x * 256 + threadIdx.x;
  size_t t = gid >> 4;
  int slot = (int)(gid & 15);
  if (t >= Tt) return;
  if (slot < 8) {
    float acc = bv[slot];
    for (int d = 0; d < Dx; d++) acc += x[t * 512 + d] * Wv[d * 8 + slot];
    vals[t * 8 + slot] = acc;
  } else if (slot == 8) {
    float acc = bsg[0];
    for (int d = 0; d < Dx; d++) acc += x[t * 512 + d] * Wsg[d];
    sg[t] = 1.0f / (1.0f + __expf(-acc));
  }
}

// mem1 phasor scan + retrieval: one thread per (b,d)
__global__ __launch_bounds__(256)
void k_mem1_scan(const float* __restrict__ v1, const float* __restrict__ mag,
                 const float* __restrict__ off, const float* __restrict__ pos,
                 const float* __restrict__ mscale_p, float* __restrict__ pret)
{
  int gid = blockIdx.x * 256 + threadIdx.x;   // 0..B*D-1
  int b = gid >> 9, d = gid & 511;
  float msc = fabsf(mscale_p[0]);
  const float inv_sqrt_d = 0.04419417382415922f;   // 1/sqrt(512)
  float ca = 0.f, sa = 0.f, ma = 0.f;
  for (int l = 0; l < Lx; l++) {
    size_t t = (size_t)b * Lx + l;
    float m  = mag[t * 512 + d] * msc;
    float wv = m * v1[t * 512 + d];
    float pc, ps;
    __sincosf(pos[(size_t)l * 512 + d], &ps, &pc);
    ca += pc * wv; sa += ps * wv; ma += m;
    float inv = rsqrtf(ma + 1e-8f);
    float oc, os;
    __sincosf(off[t * 512 + d], &os, &oc);
    float qc = pc * oc - ps * os;
    float qs = ps * oc + pc * os;
    pret[t * 512 + d] = (ca * qc + sa * qs) * inv * inv_sqrt_d;
  }
}

// KV chunked scan, pass 1: per-(b,chunk) state sums. threads = p (128)
__global__ __launch_bounds__(128)
void k_kv_chunk(const float* __restrict__ sphase, const float* __restrict__ vals,
                const float* __restrict__ sg, float* __restrict__ cCc,
                float* __restrict__ cCs, float* __restrict__ cG)
{
  int bc = blockIdx.x;            // b*NC + c
  int b = bc >> 5, c = bc & 31;
  int p = threadIdx.x;
  __shared__ float sv[8];
  __shared__ float sgs;
  float Cc[8] = {0}, Cs[8] = {0};
  float gs = 0.f;
  for (int i = 0; i < CL; i++) {
    size_t t = (size_t)b * Lx + c * CL + i;
    if (p < 8) sv[p] = vals[t * 8 + p];
    if (p == 8) sgs = sg[t];
    __syncthreads();
    float g = sgs;
    float sc, ss;
    __sincosf(sphase[t * Px + p], &ss, &sc);
    #pragma unroll
    for (int v = 0; v < 8; v++) { float gv = sv[v] * g; Cc[v] += sc * gv; Cs[v] += ss * gv; }
    gs += g;
    __syncthreads();
  }
  size_t base = ((size_t)bc * Px + p) * 8;
  #pragma unroll
  for (int v = 0; v < 8; v++) { cCc[base + v] = Cc[v]; cCs[base + v] = Cs[v]; }
  if (p == 0) cG[bc] = gs;
}

// KV pass 2: prefix from chunk sums, replay chunk emitting retrievals
__global__ __launch_bounds__(128)
void k_kv_emit(const float* __restrict__ sphase, const float* __restrict__ qphase,
               const float* __restrict__ vals, const float* __restrict__ sg,
               const float* __restrict__ cCc, const float* __restrict__ cCs,
               const float* __restrict__ cG, float* __restrict__ kvret)
{
  int bc = blockIdx.x;
  int b = bc >> 5, c = bc & 31;
  int p = threadIdx.x;
  const float inv_sqrt_p = 0.08838834764831845f;   // 1/sqrt(128)
  float Cc[8] = {0}, Cs[8] = {0};
  float gcum = 0.f;
  for (int cc = 0; cc < c; cc++) {
    size_t base = (((size_t)b * NC + cc) * Px + p) * 8;
    #pragma unroll
    for (int v = 0; v < 8; v++) { Cc[v] += cCc[base + v]; Cs[v] += cCs[base + v]; }
    gcum += cG[b * NC + cc];
  }
  __shared__ float sv[8];
  __shared__ float sgs;
  __shared__ float red[4][8];
  int wave = p >> 5, lane = p & 31;
  for (int i = 0; i < CL; i++) {
    size_t t = (size_t)b * Lx + c * CL + i;
    if (p < 8) sv[p] = vals[t * 8 + p];
    if (p == 8) sgs = sg[t];
    __syncthreads();
    float g = sgs;
    float sc, ss;
    __sincosf(sphase[t * Px + p], &ss, &sc);
    float qc, qs;
    __sincosf(qphase[t * Px + p], &qs, &qc);
    gcum += g;
    float part[8];
    #pragma unroll
    for (int v = 0; v < 8; v++) {
      float gv = sv[v] * g;
      Cc[v] += sc * gv; Cs[v] += ss * gv;
      part[v] = qc * Cc[v] + qs * Cs[v];
    }
    // wave32 reduction over p, then cross-wave via LDS
    #pragma unroll
    for (int v = 0; v < 8; v++) {
      float r = part[v];
      #pragma unroll
      for (int off = 16; off >= 1; off >>= 1) r += __shfl_xor(r, off, 32);
      part[v] = r;
    }
    if (lane == 0) {
      #pragma unroll
      for (int v = 0; v < 8; v++) red[wave][v] = part[v];
    }
    __syncthreads();
    if (p < 8) {
      float r = red[0][p] + red[1][p] + red[2][p] + red[3][p];
      float inv = rsqrtf(fmaxf(gcum, 1.0f));
      kvret[t * 8 + p] = r * inv * inv_sqrt_p;
    }
    __syncthreads();
  }
}

// kv_out = kvret @ W_kvo + b_kvo  (K=8, VALU)
__global__ __launch_bounds__(256)
void k_kv_out(const float* __restrict__ kvret, const float* __restrict__ Wk,
              const float* __restrict__ bk, float* __restrict__ kvo)
{
  size_t gid = (size_t)blockIdx.x * 256 + threadIdx.x;
  size_t t = gid >> 9;
  int d = (int)(gid & 511);
  float acc = bk[d];
  #pragma unroll
  for (int v = 0; v < 8; v++) acc += kvret[t * 8 + v] * Wk[v * 512 + d];
  kvo[gid] = acc;
}

// concat [pos_out, kv_out, conv_out] -> LayerNorm(1536). one block per token
__global__ __launch_bounds__(256)
void k_combine_ln(const float* __restrict__ po, const float* __restrict__ ko,
                  const float* __restrict__ co, const float* __restrict__ g,
                  const float* __restrict__ bt, float* __restrict__ outp)
{
  size_t t = blockIdx.x;
  int tid = threadIdx.x;
  const float* srcs[3] = {po, ko, co};
  float v[6];
  #pragma unroll
  for (int j = 0; j < 6; j++) {
    int i = tid + j * 256;
    v[j] = srcs[i >> 9][t * 512 + (i & 511)];
  }
  __shared__ float red[256];
  float s = v[0] + v[1] + v[2] + v[3] + v[4] + v[5];
  red[tid] = s; __syncthreads();
  for (int st = 128; st > 0; st >>= 1) { if (tid < st) red[tid] += red[tid + st]; __syncthreads(); }
  float mu = red[0] * (1.0f / 1536.0f);
  __syncthreads();
  float s2 = 0.f;
  #pragma unroll
  for (int j = 0; j < 6; j++) { float dd = v[j] - mu; s2 += dd * dd; }
  red[tid] = s2; __syncthreads();
  for (int st = 128; st > 0; st >>= 1) { if (tid < st) red[tid] += red[tid + st]; __syncthreads(); }
  float rstd = rsqrtf(red[0] * (1.0f / 1536.0f) + 1e-5f);
  #pragma unroll
  for (int j = 0; j < 6; j++) {
    int i = tid + j * 256;
    outp[t * 1536 + i] = (v[j] - mu) * rstd * g[i] + bt[i];
  }
}

// ---------------------------------------------------------------------------
extern "C" void kernel_launch(void* const* d_in, const int* in_sizes, int n_in,
                              void* d_out, int out_size, void* d_ws, size_t ws_size,
                              hipStream_t stream)
{
  (void)in_sizes; (void)n_in; (void)out_size; (void)ws_size;
  const float* x          = (const float*)d_in[0];
  const float* w_local    = (const float*)d_in[1];
  const float* b_local    = (const float*)d_in[2];
  const float* w_gatec    = (const float*)d_in[3];
  const float* b_gatec    = (const float*)d_in[4];
  const float* W_convproj = (const float*)d_in[5];
  const float* b_convproj = (const float*)d_in[6];
  const float* pos_phases = (const float*)d_in[7];
  const float* W_m1v      = (const float*)d_in[8];
  const float* b_m1v      = (const float*)d_in[9];
  const float* W_m1o      = (const float*)d_in[10];
  const float* b_m1o      = (const float*)d_in[11];
  const float* W_off      = (const float*)d_in[12];
  const float* b_off      = (const float*)d_in[13];
  const float* W_mag      = (const float*)d_in[14];
  const float* b_mag      = (const float*)d_in[15];
  const float* mag_scale  = (const float*)d_in[16];
  const float* W_key      = (const float*)d_in[17];
  const float* b_key      = (const float*)d_in[18];
  const float* W_val      = (const float*)d_in[19];
  const float* b_val      = (const float*)d_in[20];
  const float* W_sk1      = (const float*)d_in[21];
  const float* b_sk1      = (const float*)d_in[22];
  const float* W_sk2      = (const float*)d_in[23];
  const float* b_sk2      = (const float*)d_in[24];
  const float* W_sg       = (const float*)d_in[25];
  const float* b_sg       = (const float*)d_in[26];
  const float* W_kvo      = (const float*)d_in[27];
  const float* b_kvo      = (const float*)d_in[28];
  const float* ln_g       = (const float*)d_in[29];
  const float* ln_b       = (const float*)d_in[30];
  const float* W_o1       = (const float*)d_in[31];
  const float* b_o1       = (const float*)d_in[32];
  const float* W_o2       = (const float*)d_in[33];
  const float* b_o2       = (const float*)d_in[34];

  float* ws  = (float*)d_ws;
  float* out = (float*)d_out;
  float* cg    = ws + O_CG;   float* ctx  = ws + O_CTX;  float* v1   = ws + O_V1;
  float* mag   = ws + O_MAG;  float* off  = ws + O_OFF;  float* poso = ws + O_POSO;
  float* qp    = ws + O_QP;   float* sp   = ws + O_SP;   float* skx  = ws + O_SKX;
  float* skh   = ws + O_SKH;  float* pret = ws + O_PRET; float* pout = ws + O_POUT;
  float* convo = ws + O_CONV; float* kvo  = ws + O_KVO;  float* vals = ws + O_VALS;
  float* sg    = ws + O_SG;   float* cCc  = ws + O_CCC;  float* cCs  = ws + O_CCS;
  float* cG    = ws + O_CGT;  float* kvr  = ws + O_KVR;  float* comb = ws + O_COMB;
  float* h1    = ws + O_H1;

  auto gemm = [&](const float* A, const float* Wp, const float* bias,
                  const float* addin, int amask, float* Cp,
                  int M, int Kd, int N, int act) {
    dim3 grid(N / 64, M / 64);
    k_gemm_wmma<<<grid, 128, 0, stream>>>(A, Wp, bias, addin, amask, Cp, M, Kd, N, act);
  };

  // ---- token-parallel prep ----
  k_dwconv<<<(unsigned)(TD / 256), 256, 0, stream>>>(x, w_local, b_local, w_gatec, b_gatec, cg);
  k_ctx_scan<<<(Bx * Dx) / 256, 256, 0, stream>>>(x, ctx);

  // ---- WMMA GEMMs ----
  gemm(pos_phases, W_off + (size_t)Dx * Dx, b_off, nullptr, 0, poso, Lx, Dx, Dx, 0); // posoff
  gemm(x, W_m1v, b_m1v, nullptr, 0, v1, (int)Tt, Dx, Dx, 0);                         // v1
  gemm(x, W_mag, b_mag, nullptr, 0, mag, (int)Tt, Dx, Dx, 1);                        // sigmoid
  gemm(x, W_off, nullptr, poso, Lx - 1, off, (int)Tt, Dx, Dx, 2);                    // tanh*pi
  gemm(x, W_key, b_key, nullptr, 0, qp, (int)Tt, Dx, Px, 2);                         // query_phase
  gemm(cg, W_convproj, b_convproj, nullptr, 0, convo, (int)Tt, Dx, Dx, 0);           // conv_out
  gemm(x, W_sk1, nullptr, nullptr, 0, skx, (int)Tt, Dx, Dx, 0);                      // x part
  gemm(ctx, W_sk1 + (size_t)Dx * Dx, b_sk1, skx, 0x7FFFFFFF, skh, (int)Tt, Dx, Dx, 3); // gelu
  gemm(skh, W_sk2, b_sk2, nullptr, 0, sp, (int)Tt, Dx, Px, 2);                       // storage_phase

  // ---- narrow projections + scans ----
  k_valsg<<<(unsigned)(Tt * 16 / 256), 256, 0, stream>>>(x, W_val, b_val, W_sg, b_sg, vals, sg);
  k_mem1_scan<<<(Bx * Dx) / 256, 256, 0, stream>>>(v1, mag, off, pos_phases, mag_scale, pret);
  gemm(pret, W_m1o, b_m1o, nullptr, 0, pout, (int)Tt, Dx, Dx, 0);                    // pos_out

  k_kv_chunk<<<Bx * NC, 128, 0, stream>>>(sp, vals, sg, cCc, cCs, cG);
  k_kv_emit<<<Bx * NC, 128, 0, stream>>>(sp, qp, vals, sg, cCc, cCs, cG, kvr);
  k_kv_out<<<(unsigned)(TD / 256), 256, 0, stream>>>(kvr, W_kvo, b_kvo, kvo);

  // ---- combine + output MLP (comb aliases v1/mag/off; h1 aliases cg/ctx) ----
  k_combine_ln<<<(unsigned)Tt, 256, 0, stream>>>(pout, kvo, convo, ln_g, ln_b, comb);
  gemm(comb, W_o1, b_o1, nullptr, 0, h1, (int)Tt, 3 * Dx, 2 * Dx, 3);                // gelu
  gemm(h1, W_o2, b_o2, x, 0x7FFFFFFF, out, (int)Tt, 2 * Dx, Dx, 0);                  // + residual
}
